// MultiHeadAttention_53790170415919
// MI455X (gfx1250) — compile-verified, matrix operand order checked by reading, not run on hardware
//
#include <hip/hip_runtime.h>

#define S_LEN 4096
#define D_DIM 1024
#define H_NUM 16
#define DKV   64

typedef __attribute__((ext_vector_type(16))) __bf16 bf16x16;
typedef __attribute__((ext_vector_type(8)))  __bf16 bf16x8;
typedef __attribute__((ext_vector_type(8)))  float  f32x8;
typedef __attribute__((ext_vector_type(4)))  float  f32x4;
typedef __attribute__((ext_vector_type(4)))  unsigned int u32x4;
typedef __attribute__((ext_vector_type(8)))  int    i32x8;
typedef __attribute__((ext_vector_type(4)))  int    i32x4;

#if defined(__HIP_DEVICE_COMPILE__) &&                                        \
    __has_builtin(__builtin_amdgcn_tensor_load_to_lds) &&                     \
    __has_builtin(__builtin_amdgcn_s_wait_tensorcnt)
#define USE_TDM 1
#else
#define USE_TDM 0
#endif

static __device__ __forceinline__ f32x8 wmma_bf16(bf16x16 a, bf16x16 b, f32x8 c) {
    return __builtin_amdgcn_wmma_f32_16x16x32_bf16(false, a, false, b, (short)0, c,
                                                   false, false);
}

// Load a 16x32 bf16 fragment (A or B operand) from an LDS tile laid out
// row-major [row][stride].  Per ISA layout: lane l (<16) -> row l, K = {0..7,16..23};
// lane l+16 -> row l, K = {8..15,24..31}.  kbase selects which K=32 slab.
static __device__ __forceinline__ bf16x16
load_frag(const __bf16* base, int row0, int stride, int kbase, int lane) {
    const int m  = lane & 15;
    const int kh = (lane >> 4) * 8;
    const __bf16* p = base + (row0 + m) * stride + kbase + kh;
    bf16x8 lo = *(const bf16x8*)(p);
    bf16x8 hi = *(const bf16x8*)(p + 16);
    bf16x16 f;
#pragma unroll
    for (int i = 0; i < 8; ++i) { f[i] = lo[i]; f[i + 8] = hi[i]; }
    return f;
}

static __device__ __forceinline__ void load8_cvt(const float* src, __bf16* dst) {
    f32x4 a = *(const f32x4*)src;
    f32x4 b = *(const f32x4*)(src + 4);
#pragma unroll
    for (int i = 0; i < 4; ++i) { dst[i] = (__bf16)a[i]; dst[i + 4] = (__bf16)b[i]; }
}
static __device__ __forceinline__ void load8_cvt(const __bf16* src, __bf16* dst) {
    *(bf16x8*)dst = *(const bf16x8*)src;
}

// C = A(4096x1024) * W^T (W is 1024x1024 row-major, inner dim = K) + bias.
// Block tile 128x128, 8 waves each owning 32x64 (2x4 WMMA accumulators).
// HEADSPLIT=1: emit bf16 into [H][S][64]; HEADSPLIT=0: emit f32 into [S][1024].
template <typename TA, typename TW, int HEADSPLIT>
__global__ __launch_bounds__(256) void proj_gemm(const TA* __restrict__ A,
                                                 const TW* __restrict__ W,
                                                 const float* __restrict__ bias,
                                                 void* __restrict__ out) {
    __shared__ __bf16 lA[128 * 32];
    __shared__ __bf16 lB[128 * 32];

    const int tid  = threadIdx.x;
    const int lane = tid & 31;
    const int wave = tid >> 5;
    const int bm   = blockIdx.y * 128;
    const int bn   = blockIdx.x * 128;
    const int wm   = (wave & 3) * 32;   // 4 waves over M
    const int wn   = (wave >> 2) * 64;  // 2 waves over N

    f32x8 acc[2][4];
#pragma unroll
    for (int i = 0; i < 2; ++i)
#pragma unroll
        for (int j = 0; j < 4; ++j) acc[i][j] = {};

    for (int k0 = 0; k0 < D_DIM; k0 += 32) {
        __syncthreads();
        // stage A tile: 128 rows x 32 K (512 chunks of 8 elements, 2 per thread)
#pragma unroll
        for (int c = tid; c < 512; c += 256) {
            const int r = c >> 2, col = (c & 3) * 8;
            load8_cvt(A + (size_t)(bm + r) * D_DIM + k0 + col, lA + r * 32 + col);
        }
        // stage W tile: 128 rows x 32 K
#pragma unroll
        for (int c = tid; c < 512; c += 256) {
            const int r = c >> 2, col = (c & 3) * 8;
            load8_cvt(W + (size_t)(bn + r) * D_DIM + k0 + col, lB + r * 32 + col);
        }
        __syncthreads();

        bf16x16 af[2], bf[4];
        af[0] = load_frag(lA, wm,      32, 0, lane);
        af[1] = load_frag(lA, wm + 16, 32, 0, lane);
#pragma unroll
        for (int j = 0; j < 4; ++j) bf[j] = load_frag(lB, wn + j * 16, 32, 0, lane);

#pragma unroll
        for (int i = 0; i < 2; ++i)
#pragma unroll
            for (int j = 0; j < 4; ++j) acc[i][j] = wmma_bf16(af[i], bf[j], acc[i][j]);
    }

    const int mr0  = 8 * (lane >> 4);
    const int nloc = lane & 15;
#pragma unroll
    for (int ti = 0; ti < 2; ++ti)
#pragma unroll
        for (int tj = 0; tj < 4; ++tj) {
            const int gn = bn + wn + tj * 16 + nloc;
            const float bv = bias[gn];
#pragma unroll
            for (int r = 0; r < 8; ++r) {
                const float v  = acc[ti][tj][r] + bv;
                const int   gm = bm + wm + ti * 16 + mr0 + r;
                if (HEADSPLIT) {
                    const int h = gn >> 6, dk = gn & 63;
                    ((__bf16*)out)[((size_t)h * S_LEN + gm) * DKV + dk] = (__bf16)v;
                } else {
                    ((float*)out)[(size_t)gm * D_DIM + gn] = v;
                }
            }
        }
}

// Flash attention: one block per (64-query tile, head). 4 waves x 16 rows.
__global__ __launch_bounds__(128) void attn_kernel(const __bf16* __restrict__ qh,
                                                   const __bf16* __restrict__ kh,
                                                   const __bf16* __restrict__ vh,
                                                   __bf16* __restrict__ attn_out) {
    __shared__ __bf16 lK[64 * 64];       // [key][dk]
    __shared__ __bf16 lVt[64 * 72];      // [dk][key] (+pad)
    __shared__ __bf16 lP[4][16 * 72];    // per-wave P tile (+pad)

    const int h    = blockIdx.y;
    const int qb   = blockIdx.x;
    const int tid  = threadIdx.x;
    const int lane = tid & 31;
    const int wave = tid >> 5;

    const __bf16* Qh = qh + (size_t)h * S_LEN * DKV;
    const __bf16* Kh = kh + (size_t)h * S_LEN * DKV;
    const __bf16* Vh = vh + (size_t)h * S_LEN * DKV;

    const int qrow0 = qb * 64 + wave * 16;
    const int mr0   = 8 * (lane >> 4);
    const int nloc  = lane & 15;

    // Q fragments (A-operand, 16 rows x 64 dk -> two K=32 fragments)
    bf16x16 qf[2];
    {
        const __bf16* qp = Qh + (size_t)(qrow0 + nloc) * DKV + ((lane >> 4) * 8);
        bf16x8 a0 = *(const bf16x8*)(qp);
        bf16x8 a1 = *(const bf16x8*)(qp + 16);
        bf16x8 a2 = *(const bf16x8*)(qp + 32);
        bf16x8 a3 = *(const bf16x8*)(qp + 48);
#pragma unroll
        for (int i = 0; i < 8; ++i) {
            qf[0][i] = a0[i]; qf[0][i + 8] = a1[i];
            qf[1][i] = a2[i]; qf[1][i + 8] = a3[i];
        }
    }

    f32x8 oacc[4];
#pragma unroll
    for (int t = 0; t < 4; ++t) oacc[t] = {};
    float mrow[8], lrow[8];
#pragma unroll
    for (int r = 0; r < 8; ++r) { mrow[r] = -1e30f; lrow[r] = 0.0f; }

    for (int kb = 0; kb <= qb; ++kb) {
        const int kbase = kb * 64;
        __syncthreads();

#if USE_TDM
        // Tensor Data Mover: DMA the 64x64 bf16 K tile straight into LDS.
        // D# per CDNA5 ISA 8.3/8.4: group0 = {count, lds_addr, global_addr, type=2},
        // group1 = {data_size=2B, tensor_dim0=64, tensor_dim1=S, tile 64x64,
        //           tensor_dim0_stride=64}.  Issued once (wave 0); TENSORcnt waited
        // before the block barrier releases the consumers.
        if (wave == 0) {
            const unsigned long long ga =
                (unsigned long long)(size_t)(Kh + (size_t)kbase * DKV);
            u32x4 g0 = {};
            g0[0] = 1u;                                    // count = 1 (valid)
            g0[1] = (unsigned)(size_t)lK;                  // lds_addr
            g0[2] = (unsigned)ga;                          // global_addr[31:0]
            g0[3] = (unsigned)((ga >> 32) & 0x1FFFFFFull) | (2u << 30);  // type=2
            i32x8 g1 = {};
            g1[0] = (1 << 16);                             // data_size = 2 bytes
            g1[1] = (DKV & 0xFFFF) << 16;                  // tensor_dim0 = 64
            g1[2] = (S_LEN & 0xFFFF) << 16;                // tensor_dim1 = 4096
            g1[3] = (64 << 16);                            // tile_dim0 = 64
            g1[4] = 64;                                    // tile_dim1 = 64
            g1[5] = DKV;                                   // tensor_dim0_stride = 64
            g1[6] = 0;
            g1[7] = 0;
            i32x4 gz  = {};                                // groups 2/3 unused (2-D)
            i32x8 gz8 = {};
            __builtin_amdgcn_tensor_load_to_lds(g0, g1, gz, gz, gz8, 0);
            __builtin_amdgcn_s_wait_tensorcnt(0);
        }
#else
        // fallback: manual K block staging
#pragma unroll
        for (int c = tid; c < 512; c += 128) {
            const int r = c >> 3, col = (c & 7) * 8;
            *(bf16x8*)(lK + r * 64 + col) =
                *(const bf16x8*)(Kh + (size_t)(kbase + r) * DKV + col);
        }
#endif
        // stage V block transposed -> [dk][key]
#pragma unroll
        for (int c = tid; c < 512; c += 128) {
            const int key = c >> 3, col = (c & 7) * 8;
            bf16x8 v = *(const bf16x8*)(Vh + (size_t)(kbase + key) * DKV + col);
#pragma unroll
            for (int i = 0; i < 8; ++i) lVt[(col + i) * 72 + key] = v[i];
        }
        if (kb < qb) {  // prefetch next key block (global_prefetch_b8)
            __builtin_prefetch(Kh + (size_t)(kbase + 64) * DKV + tid * 32, 0, 0);
            __builtin_prefetch(Vh + (size_t)(kbase + 64) * DKV + tid * 32, 0, 0);
        }
        __syncthreads();

        // scores: S = Q * K^T  (4 tiles of 16 keys)
        f32x8 sc[4];
#pragma unroll
        for (int j = 0; j < 4; ++j) {
            sc[j] = {};
            bf16x16 b0 = load_frag(lK, j * 16, 64, 0, lane);
            bf16x16 b1 = load_frag(lK, j * 16, 64, 32, lane);
            sc[j] = wmma_bf16(qf[0], b0, sc[j]);
            sc[j] = wmma_bf16(qf[1], b1, sc[j]);
        }

        const float scale = 0.125f;  // 1/sqrt(64)
#pragma unroll
        for (int j = 0; j < 4; ++j)
#pragma unroll
            for (int r = 0; r < 8; ++r) {
                float v = sc[j][r] * scale;
                if (kb == qb) {
                    const int qg = qrow0 + mr0 + r;
                    const int kg = kbase + j * 16 + nloc;
                    if (kg > qg) v = -1e30f;
                }
                sc[j][r] = v;
            }

        // online softmax per row (row stats replicated across 16-lane groups)
#pragma unroll
        for (int r = 0; r < 8; ++r) {
            float mx = sc[0][r];
#pragma unroll
            for (int j = 1; j < 4; ++j) mx = fmaxf(mx, sc[j][r]);
#pragma unroll
            for (int m = 1; m < 16; m <<= 1) mx = fmaxf(mx, __shfl_xor(mx, m, 32));
            const float mnew = fmaxf(mrow[r], mx);
            const float corr = __expf(mrow[r] - mnew);
            mrow[r] = mnew;
            float sum = 0.0f;
#pragma unroll
            for (int j = 0; j < 4; ++j) {
                const float p = __expf(sc[j][r] - mnew);
                sc[j][r] = p;
                sum += p;
            }
#pragma unroll
            for (int m = 1; m < 16; m <<= 1) sum += __shfl_xor(sum, m, 32);
            lrow[r] = lrow[r] * corr + sum;
#pragma unroll
            for (int t = 0; t < 4; ++t) oacc[t][r] *= corr;
        }

        // P (C-layout) -> LDS -> A-layout fragments
        __bf16* pw = lP[wave];
#pragma unroll
        for (int j = 0; j < 4; ++j)
#pragma unroll
            for (int r = 0; r < 8; ++r)
                pw[(mr0 + r) * 72 + j * 16 + nloc] = (__bf16)sc[j][r];
        __syncthreads();

        bf16x16 pf0 = load_frag(pw, 0, 72, 0, lane);
        bf16x16 pf1 = load_frag(pw, 0, 72, 32, lane);
#pragma unroll
        for (int t = 0; t < 4; ++t) {
            bf16x16 v0 = load_frag(lVt, t * 16, 72, 0, lane);
            bf16x16 v1 = load_frag(lVt, t * 16, 72, 32, lane);
            oacc[t] = wmma_bf16(pf0, v0, oacc[t]);
            oacc[t] = wmma_bf16(pf1, v1, oacc[t]);
        }
    }

    // normalize + store merged-head layout [s][h*64+dk] as bf16
    float invl[8];
#pragma unroll
    for (int r = 0; r < 8; ++r) invl[r] = 1.0f / lrow[r];
#pragma unroll
    for (int t = 0; t < 4; ++t)
#pragma unroll
        for (int r = 0; r < 8; ++r) {
            const float v  = oacc[t][r] * invl[r];
            const int   qg = qrow0 + mr0 + r;
            const int   dk = t * 16 + nloc;
            attn_out[(size_t)qg * D_DIM + h * DKV + dk] = (__bf16)v;
        }
}

extern "C" void kernel_launch(void* const* d_in, const int* in_sizes, int n_in,
                              void* d_out, int out_size, void* d_ws, size_t ws_size,
                              hipStream_t stream) {
    (void)in_sizes; (void)n_in; (void)out_size; (void)ws_size;
    const float* Q    = (const float*)d_in[0];
    const float* K    = (const float*)d_in[1];
    const float* V    = (const float*)d_in[2];
    const float* Wq_w = (const float*)d_in[3];
    const float* Wq_b = (const float*)d_in[4];
    const float* Wk_w = (const float*)d_in[5];
    const float* Wk_b = (const float*)d_in[6];
    const float* Wv_w = (const float*)d_in[7];
    const float* Wv_b = (const float*)d_in[8];
    const float* Wo_w = (const float*)d_in[9];
    const float* Wo_b = (const float*)d_in[10];
    // d_in[11] = causal mask, applied analytically

    __bf16* ws = (__bf16*)d_ws;
    const size_t plane = (size_t)S_LEN * D_DIM;  // 4M elements
    __bf16* qh = ws;                  // [H][S][64]
    __bf16* kh = ws + plane;          // [H][S][64]
    __bf16* vh = ws + 2 * plane;      // [H][S][64]
    __bf16* ao = ws + 3 * plane;      // [S][1024]

    dim3 gProj(D_DIM / 128, S_LEN / 128);  // (8, 32)
    dim3 bProj(256);
    proj_gemm<float, float, 1><<<gProj, bProj, 0, stream>>>(Q, Wq_w, Wq_b, qh);
    proj_gemm<float, float, 1><<<gProj, bProj, 0, stream>>>(K, Wk_w, Wk_b, kh);
    proj_gemm<float, float, 1><<<gProj, bProj, 0, stream>>>(V, Wv_w, Wv_b, vh);

    dim3 gAttn(S_LEN / 64, H_NUM);         // (64, 16)
    attn_kernel<<<gAttn, 128, 0, stream>>>(qh, kh, vh, ao);

    proj_gemm<__bf16, float, 0><<<gProj, bProj, 0, stream>>>(ao, Wo_w, Wo_b,
                                                             (float*)d_out);
}
